// Model_53266184405577
// MI455X (gfx1250) — 8-bit matrix operand order repaired and checked by reading, compiled, not run on hardware
//
#include <hip/hip_runtime.h>
#include <stdint.h>


typedef __attribute__((ext_vector_type(8))) int v8i;

#define EPS 1e-4f

__global__ __launch_bounds__(256) void prep_weights_kernel(
    const float* __restrict__ w, int8_t* __restrict__ w8,
    int Cout, int Cin, int Mpad, int tflag) {
  const int K = Cin * 9;
  const size_t total = (size_t)Mpad * (size_t)K;
  size_t i = (size_t)blockIdx.x * blockDim.x + threadIdx.x;
  if (i >= total) return;
  int m = (int)(i / (size_t)K);
  int k = (int)(i - (size_t)m * (size_t)K);
  int8_t v = 0;
  if (m < Cout) {
    int ci = k / 9;
    int r  = k - ci * 9;
    int ky = r / 3;
    int kx = r - ky * 3;
    float wv;
    if (tflag)
      wv = w[(((size_t)ci * Cout + m) * 3 + (2 - ky)) * 3 + (2 - kx)];
    else
      wv = w[(((size_t)m * Cin + ci) * 3 + ky) * 3 + kx];
    v = (wv >= 0.0f) ? (int8_t)1 : (int8_t)-1;
  }
  w8[i] = v;
}

__global__ __launch_bounds__(256) void conv0_kernel(
    const float* __restrict__ x, const float* __restrict__ w,
    const float* __restrict__ b, float* __restrict__ out) {
  const size_t total = (size_t)8 * 64 * 256 * 256;
  size_t idx = (size_t)blockIdx.x * blockDim.x + threadIdx.x;
  if (idx >= total) return;
  int ox = (int)(idx & 255);
  int oy = (int)((idx >> 8) & 255);
  int co = (int)((idx >> 16) & 63);
  int n  = (int)(idx >> 22);
  float acc = b[co];
#pragma unroll
  for (int ci = 0; ci < 3; ++ci)
#pragma unroll
    for (int ky = 0; ky < 3; ++ky) {
      int iy = oy + ky - 1;
      if (iy < 0 || iy > 255) continue;
#pragma unroll
      for (int kx = 0; kx < 3; ++kx) {
        int ix = ox + kx - 1;
        if (ix < 0 || ix > 255) continue;
        acc += x[(((size_t)n * 3 + ci) * 256 + iy) * 256 + ix] *
               w[(((size_t)co * 3 + ci) * 3 + ky) * 3 + kx];
      }
    }
  out[idx] = acc;
}

#define CHK 576

__global__ __launch_bounds__(128) void bconv_wmma_kernel(
    const int8_t* __restrict__ act, const int8_t* __restrict__ w8,
    const float* __restrict__ bias, float* __restrict__ out,
    int Cin, int Cout, int Hin, int Win, int Hout, int Wout, int mode) {
  __shared__ int8_t lds[64 * CHK];
  const int K     = Cin * 9;
  const int n     = blockIdx.z;
  const int mbase = blockIdx.y * 64;
  const int pixb  = blockIdx.x * 64;
  const int HWout = Hout * Wout;
  const int tid   = threadIdx.x;

  const int wave = tid >> 5;
  const int lane = tid & 31;
  const int half = lane >> 4;
  const int l15  = lane & 15;
  const int mrow = mbase + wave * 16 + l15;

  const int fcol  = tid >> 1;
  const int fci0  = tid & 1;
  const int fpix  = pixb + fcol;
  const bool fval = fpix < HWout;
  const int foy   = fval ? (fpix / Wout) : 0;
  const int fox   = fval ? (fpix - foy * Wout) : 0;

  v8i acc0 = {0, 0, 0, 0, 0, 0, 0, 0};
  v8i acc1 = acc0, acc2 = acc0, acc3 = acc0;

  const int nchunks = Cin >> 6;
  for (int chunk = 0; chunk < nchunks; ++chunk) {
    const int cibase = chunk << 6;
    __syncthreads();

    for (int cil = fci0; cil < 64; cil += 2) {
      const int ci = cibase + cil;
      const size_t abase = ((size_t)n * Cin + ci) * (size_t)Hin * Win;
      int8_t* dst = lds + fcol * CHK + cil * 9;
      if (mode == 2) {
#pragma unroll
        for (int r = 0; r < 9; ++r) {
          const int ky = r / 3, kx = r % 3;
          int8_t v = 0;
          if (fval) {
            int ty = foy + ky - 1, tx = fox + kx - 1;
            if (ty >= 0 && tx >= 0 && !(ty & 1) && !(tx & 1)) {
              int iy = ty >> 1, ix = tx >> 1;
              if (iy < Hin && ix < Win) v = act[abase + (size_t)iy * Win + ix];
            }
          }
          dst[r] = v;
        }
      } else {
        const int s = (mode == 1) ? 2 : 1;
#pragma unroll
        for (int r = 0; r < 9; ++r) {
          const int ky = r / 3, kx = r % 3;
          int8_t v = 0;
          if (fval) {
            int iy = foy * s + ky - 1, ix = fox * s + kx - 1;
            if (iy >= 0 && iy < Hin && ix >= 0 && ix < Win)
              v = act[abase + (size_t)iy * Win + ix];
          }
          dst[r] = v;
        }
      }
    }
    __syncthreads();

    const int kchunk = chunk * CHK;
#pragma unroll
    for (int ks = 0; ks < 9; ++ks) {
      const int kb = ks << 6;
      const int* wp = (const int*)(w8 + (size_t)mrow * K + kchunk + kb);
      __builtin_prefetch(wp + 16, 0, 1);
      v8i a;
#pragma unroll
      for (int v = 0; v < 8; ++v)
        a[v] = wp[((v >> 2) << 3) + (v & 3) + (half << 2)];
#pragma unroll
      for (int t = 0; t < 4; ++t) {
        const int8_t* bp = lds + (t * 16 + l15) * CHK + kb + (half << 4);
        v8i bfrag;
#pragma unroll
        for (int v = 0; v < 8; ++v)
          bfrag[v] = *(const int*)(bp + ((v >> 2) << 5) + ((v & 3) << 2));
        v8i* pacc = (t == 0) ? &acc0 : (t == 1) ? &acc1 : (t == 2) ? &acc2 : &acc3;
        *pacc = __builtin_amdgcn_wmma_i32_16x16x64_iu8(
             true, a,  true, bfrag, *pacc,
             false,  false);
      }
    }
  }

#pragma unroll
  for (int t = 0; t < 4; ++t) {
    const v8i& acc = (t == 0) ? acc0 : (t == 1) ? acc1 : (t == 2) ? acc2 : acc3;
    const int pix = pixb + t * 16 + l15;
    if (pix < HWout) {
#pragma unroll
      for (int r = 0; r < 8; ++r) {
        int co = mbase + wave * 16 + r + 8 * half;
        if (co < Cout)
          out[((size_t)n * Cout + co) * (size_t)HWout + pix] =
              (float)acc[r] + bias[co];
      }
    }
  }
}

__global__ __launch_bounds__(256) void stats_kernel(
    const float* __restrict__ y, float* __restrict__ stats,
    int C, int HW, int N) {
  const int c = blockIdx.x;
  float s = 0.0f, ss = 0.0f;
  for (int nimg = 0; nimg < N; ++nimg) {
    const float* p = y + ((size_t)nimg * C + c) * (size_t)HW;
    for (int rr = threadIdx.x; rr < HW; rr += 256) {
      float v = p[rr];
      s += v;
      ss += v * v;
    }
  }
  __shared__ float shs[256], shq[256];
  shs[threadIdx.x] = s;
  shq[threadIdx.x] = ss;
  __syncthreads();
  for (int o = 128; o > 0; o >>= 1) {
    if (threadIdx.x < (unsigned)o) {
      shs[threadIdx.x] += shs[threadIdx.x + o];
      shq[threadIdx.x] += shq[threadIdx.x + o];
    }
    __syncthreads();
  }
  if (threadIdx.x == 0) {
    float per = (float)N * (float)HW;
    float m   = shs[0] / per;
    float var = shq[0] / per - m * m;
    stats[2 * c]     = m;
    stats[2 * c + 1] = rsqrtf(var + EPS);
  }
}

__global__ __launch_bounds__(256) void bnact_kernel(
    const float* __restrict__ y, const float* __restrict__ stats,
    const float* __restrict__ g, const float* __restrict__ be,
    int8_t* __restrict__ act, float* __restrict__ fout,
    int C, int HW) {
  const int plane = blockIdx.y;
  const int c     = plane % C;
  const int hw    = blockIdx.x * 256 + threadIdx.x;
  if (hw >= HW) return;
  const size_t i = (size_t)plane * (size_t)HW + hw;
  const float sc = stats[2 * c + 1] * g[c];
  float v = (y[i] - stats[2 * c]) * sc + be[c];
  if (act)
    act[i] = (v >= 0.0f) ? (int8_t)1 : (int8_t)-1;
  else
    fout[i] = v;
}

static inline size_t cdiv(size_t a, size_t b) { return (a + b - 1) / b; }

extern "C" void kernel_launch(void* const* d_in, const int* in_sizes, int n_in,
                              void* d_out, int out_size, void* d_ws, size_t ws_size,
                              hipStream_t stream) {
  (void)in_sizes; (void)n_in; (void)out_size; (void)ws_size;
  char* ws = (char*)d_ws;

  float*  convf = (float*)ws;
  int8_t* actA  = (int8_t*)(ws + 134217728ull);
  int8_t* actB  = (int8_t*)(ws + 167772160ull);
  int8_t* wbase = (int8_t*)(ws + 201326592ull);
  float*  stats = (float*)(ws + 203423744ull);

  int8_t* W1  = wbase + 0;
  int8_t* W2  = wbase + 36864;
  int8_t* W3  = wbase + 110592;
  int8_t* W4  = wbase + 258048;
  int8_t* W5  = wbase + 552960;
  int8_t* WT0 = wbase + 1142784;
  int8_t* W6  = wbase + 1437696;
  int8_t* WT1 = wbase + 1585152;
  int8_t* W7  = wbase + 1658880;
  int8_t* W8c = wbase + 1695744;

  const float* in[45];
  for (int i = 0; i < 45; ++i) in[i] = (const float*)d_in[i];

  auto prep = [&](const float* w, int8_t* w8, int Cout, int Cin, int Mpad, int tflag) {
    size_t tot = (size_t)Mpad * (size_t)(Cin * 9);
    prep_weights_kernel<<<(unsigned)cdiv(tot, 256), 256, 0, stream>>>(
        w, w8, Cout, Cin, Mpad, tflag);
  };
  prep(in[3],  W1,  64,  64,  64,  0);
  prep(in[5],  W2,  128, 64,  128, 0);
  prep(in[7],  W3,  128, 128, 128, 0);
  prep(in[9],  W4,  256, 128, 256, 0);
  prep(in[11], W5,  256, 256, 256, 0);
  prep(in[13], WT0, 128, 256, 128, 1);
  prep(in[15], W6,  128, 128, 128, 0);
  prep(in[17], WT1, 64,  128, 64,  1);
  prep(in[19], W7,  64,  64,  64,  0);
  prep(in[21], W8c, 3,   64,  64,  0);

  const int N = 8;

  {
    size_t tot = (size_t)N * 64 * 256 * 256;
    conv0_kernel<<<(unsigned)cdiv(tot, 256), 256, 0, stream>>>(
        in[0], in[1], in[2], convf);
    stats_kernel<<<64, 256, 0, stream>>>(convf, stats, 64, 65536, N);
    dim3 bg((unsigned)cdiv(65536, 256), N * 64);
    bnact_kernel<<<bg, 256, 0, stream>>>(
        convf, stats, in[23], in[24], actA, nullptr, 64, 65536);
  }

  auto run = [&](const int8_t* src, int8_t* dst, const int8_t* w8,
                 const float* bias, const float* g, const float* be,
                 int Cin, int Cout, int Hin, int Win, int Hout, int Wout,
                 int mode, bool final_layer) {
    int HW = Hout * Wout;
    dim3 grid((unsigned)cdiv(HW, 64), (unsigned)cdiv(Cout, 64), N);
    bconv_wmma_kernel<<<grid, 128, 0, stream>>>(
        src, w8, bias, convf, Cin, Cout, Hin, Win, Hout, Wout, mode);
    stats_kernel<<<Cout, 256, 0, stream>>>(convf, stats, Cout, HW, N);
    dim3 bg((unsigned)cdiv(HW, 256), N * Cout);
    bnact_kernel<<<bg, 256, 0, stream>>>(
        convf, stats, g, be,
        final_layer ? nullptr : dst,
        final_layer ? (float*)d_out : nullptr,
        Cout, HW);
  };

  run(actA, actB, W1,  in[4],  in[25], in[26], 64,  64,  256, 256, 256, 256, 0, false);
  run(actB, actA, W2,  in[6],  in[27], in[28], 64,  128, 256, 256, 128, 128, 1, false);
  run(actA, actB, W3,  in[8],  in[29], in[30], 128, 128, 128, 128, 128, 128, 0, false);
  run(actB, actA, W4,  in[10], in[31], in[32], 128, 256, 128, 128, 64,  64,  1, false);
  run(actA, actB, W5,  in[12], in[33], in[34], 256, 256, 64,  64,  64,  64,  0, false);
  run(actB, actA, WT0, in[14], in[35], in[36], 256, 128, 64,  64,  127, 127, 2, false);
  run(actA, actB, W6,  in[16], in[37], in[38], 128, 128, 127, 127, 127, 127, 0, false);
  run(actB, actA, WT1, in[18], in[39], in[40], 128, 64,  127, 127, 253, 253, 2, false);
  run(actA, actB, W7,  in[20], in[41], in[42], 64,  64,  253, 253, 253, 253, 0, false);
  run(actB, actA, W8c, in[22], in[43], in[44], 64,  3,   253, 253, 253, 253, 0, true);
}
